// MoEVelocityNet_12713103196698
// MI455X (gfx1250) — matrix operand order repaired and checked by reading, not compiled, not run on hardware
//
#include <hip/hip_runtime.h>
#include <hip/hip_bf16.h>

// ---------------- problem constants ----------------
#define BTOK   65536
#define EMBED  256
#define CONDD  256
#define TIMED  128
#define HID    256
#define NEXP   8
#define RHID   128
#define IN_DIM 640
#define MTILE  64          // tokens per workgroup

// ---------------- vector types (raw clang ext vectors, union-safe) ----------
typedef __attribute__((ext_vector_type(16))) __bf16        v16bf;
typedef __attribute__((ext_vector_type(8)))  float         v8f;
typedef __attribute__((ext_vector_type(4)))  unsigned int  u32x4;
typedef __attribute__((ext_vector_type(4)))  float         f32x4;

union Frag {
    u32x4 u[2];   // 32 bytes
    v16bf v;      // 16 bf16 = 32 bytes
};

// ---------------- LDS layout (bytes) ----------------
#define XS_LD    648                     // bf16 stride for x tile (640 + 8 pad)
#define H_LD     264                     // bf16 stride for h tiles (256 + 8 pad)
#define WST_LD   40                      // bf16 stride in weight-stage chunk (32 + 8 pad; 80B rows, 16B-aligned)
#define XS_OFF   0                       // bf16 [64][648]  = 82944 B
#define H1_OFF   82944                   // bf16 [64][264]  = 33792 B
#define H2_OFF   116736                  // bf16 [64][264]  = 33792 B
#define VEL_OFF  150528                  // f32  [64][256]  = 65536 B
#define GATE_OFF 216064                  // f32  [64][8]    =  2048 B
#define WST_OFF  218112                  // bf16 3 x [256][40] = 3 x 20480 B (triple buffer)
#define WCH_BYTES 20480
#define SMEM_BYTES 279552

__device__ __forceinline__ float silu_f(float x) { return x / (1.0f + __expf(-x)); }

__device__ __forceinline__ unsigned lds_addr_of(const void* p) {
    // LDS aperture: flat addr bits [31:0] are the LDS byte offset (ISA 10.2)
    return (unsigned)(size_t)p;
}

__device__ __forceinline__ void async_wait0() {
#if __has_builtin(__builtin_amdgcn_s_wait_asynccnt)
    __builtin_amdgcn_s_wait_asynccnt(0);
#else
    asm volatile("s_wait_asynccnt 0" ::: "memory");
#endif
}
__device__ __forceinline__ void async_wait4() {
#if __has_builtin(__builtin_amdgcn_s_wait_asynccnt)
    __builtin_amdgcn_s_wait_asynccnt(4);
#else
    asm volatile("s_wait_asynccnt 4" ::: "memory");
#endif
}

// Async-copy one [256 N][32 K] bf16 chunk (col-major rows of 64B) into LDS.
// 1024 x 16B transfers; 4 per thread; per-wave ASYNCcnt += 4.
__device__ __forceinline__ void stage_chunk_async(const __hip_bfloat16* src_base, int KTOT,
                                                  int k0, unsigned lds_base, int tid) {
#pragma unroll
    for (int q4 = 0; q4 < 4; ++q4) {
        int c = tid * 4 + q4;          // 0..1023
        int n = c >> 2, q = c & 3;
        unsigned long long g =
            (unsigned long long)(size_t)(src_base + (size_t)n * KTOT + k0 + q * 8);
        unsigned l = lds_base + (unsigned)(n * WST_LD + q * 8) * 2u;
        asm volatile("global_load_async_to_lds_b128 %0, %1, off"
                     :: "v"(l), "v"(g) : "memory");
    }
}

// Load a 16x32 bf16 A-fragment from LDS (row-major, stride `ld` halves).
// lanes 0-15 -> M=lane, K {0..15}; lanes 16-31 -> M=lane-16, K {16..31} (matches B fragments).
__device__ __forceinline__ v16bf lds_a_frag(const __hip_bfloat16* base, int ld,
                                            int r0, int k0, int lane) {
    int row = r0 + (lane & 15);
    int kb  = k0 + ((lane >> 4) << 4);
    Frag f;
    f.u[0] = *(const u32x4*)(base + row * ld + kb);
    f.u[1] = *(const u32x4*)(base + row * ld + kb + 8);
    return f.v;
}

// 32x16 bf16 B-fragment from global, weights col-major [N][K].
__device__ __forceinline__ v16bf glb_b_frag(const __hip_bfloat16* w, int K,
                                            int n0, int k0, int lane) {
    int col = n0 + (lane & 15);
    int kb  = k0 + ((lane >> 4) << 4);
    const u32x4* p = (const u32x4*)(w + (size_t)col * K + kb);
    Frag f;
    f.u[0] = p[0];
    f.u[1] = p[1];
    return f.v;
}

// 32x16 bf16 B-fragment from a staged LDS chunk ([N][WST_LD], k relative to chunk).
__device__ __forceinline__ v16bf lds_b_frag(const __hip_bfloat16* buf, int n0, int lane) {
    int col = n0 + (lane & 15);
    int kb  = (lane >> 4) << 4;
    const u32x4* p = (const u32x4*)(buf + col * WST_LD + kb);
    Frag f;
    f.u[0] = p[0];
    f.u[1] = p[1];
    return f.v;
}

// Direct-B GEMM slice (B from global; used for small router/time GEMMs).
template <int KTOT, int LDA, int NT>
__device__ __forceinline__ void gemm_slice(const __hip_bfloat16* As, int r0,
                                           const __hip_bfloat16* Bw, int n0,
                                           int lane, v8f* acc) {
#pragma unroll
    for (int j = 0; j < NT; ++j)
#pragma unroll
        for (int r = 0; r < 8; ++r) acc[j][r] = 0.0f;

    for (int k0 = 0; k0 < KTOT; k0 += 32) {
        v16bf a = lds_a_frag(As, LDA, r0, k0, lane);
#pragma unroll
        for (int j = 0; j < NT; ++j) {
            v16bf b = glb_b_frag(Bw, KTOT, n0 + 16 * j, k0, lane);
            acc[j] = __builtin_amdgcn_wmma_f32_16x16x32_bf16(
                false, a, false, b, (short)0, acc[j], false, false);
        }
    }
}

// Staged-B GEMM slice: B chunks triple-buffered in LDS via async-to-LDS.
// Prefetch distance = 2 chunks; threshold wait (ASYNCcnt <= 4) keeps the newest
// stage in flight while guaranteeing the chunk needed next is resident
// (each stage = exactly 4 async ops per wave; async loads complete in order).
// Collective across the workgroup.
template <int KTOT, int LDA, int NT>
__device__ __forceinline__ void gemm_staged(const __hip_bfloat16* As, int r0,
                                            const __hip_bfloat16* Bw, int n0,
                                            int tid, int lane, v8f* acc,
                                            __hip_bfloat16* wst) {
    constexpr int NC = KTOT / 32;  // >= 2 for all uses
#pragma unroll
    for (int j = 0; j < NT; ++j)
#pragma unroll
        for (int r = 0; r < 8; ++r) acc[j][r] = 0.0f;

    const unsigned wbase = lds_addr_of(wst);
    stage_chunk_async(Bw, KTOT, 0,  wbase, tid);
    stage_chunk_async(Bw, KTOT, 32, wbase + WCH_BYTES, tid);
    async_wait4();          // chunk 0 resident, chunk 1 may be in flight
    __syncthreads();

    for (int i = 0; i < NC; ++i) {
        const int  k0     = i * 32;
        const bool issued = (i + 2 < NC);
        if (issued)
            stage_chunk_async(Bw, KTOT, k0 + 64,
                              wbase + (unsigned)(((i + 2) % 3) * WCH_BYTES), tid);

        const __hip_bfloat16* cur =
            (const __hip_bfloat16*)((const unsigned char*)wst + (i % 3) * WCH_BYTES);
        v16bf a = lds_a_frag(As, LDA, r0, k0, lane);
#pragma unroll
        for (int j = 0; j < NT; ++j) {
            v16bf b = lds_b_frag(cur, n0 + 16 * j, lane);
            acc[j] = __builtin_amdgcn_wmma_f32_16x16x32_bf16(
                false, a, false, b, (short)0, acc[j], false, false);
        }
        if (issued) async_wait4();   // chunk i+1 resident, i+2 may fly
        else        async_wait0();   // tail: everything resident
        __syncthreads();
    }
}

// ---------------- weight prep: f32 [.,K,N] -> bf16 col-major [.,N,K] ----------------
__global__ void cvt_transpose(const float* __restrict__ in,
                              __hip_bfloat16* __restrict__ out,
                              int K, int N, long total) {
    long idx = (long)blockIdx.x * blockDim.x + threadIdx.x;
    if (idx >= total) return;
    long kn = (long)K * N;
    long e  = idx / kn;
    long r  = idx - e * kn;
    int  k  = (int)(r / N);
    int  n  = (int)(r - (long)k * N);
    out[e * kn + (long)n * K + k] = __float2bfloat16(in[idx]);
}

// ---------------- fused MoE velocity kernel ----------------
__global__ __launch_bounds__(256, 1)
void moe_main(const float* __restrict__ z_t, const float* __restrict__ t,
              const float* __restrict__ cond,
              const float* __restrict__ tb1, const float* __restrict__ tb2,
              const float* __restrict__ eb1, const float* __restrict__ eb2,
              const float* __restrict__ eb3,
              const float* __restrict__ rb1, const float* __restrict__ rW2,
              const float* __restrict__ rb2,
              const __hip_bfloat16* __restrict__ w1t,
              const __hip_bfloat16* __restrict__ w2t,
              const __hip_bfloat16* __restrict__ w3t,
              const __hip_bfloat16* __restrict__ r1t,
              const __hip_bfloat16* __restrict__ t1t,
              const __hip_bfloat16* __restrict__ t2t,
              float* __restrict__ out_vel, float* __restrict__ out_gate) {
    __shared__ __align__(16) unsigned char smem[SMEM_BYTES];
    __hip_bfloat16* xs   = (__hip_bfloat16*)(smem + XS_OFF);
    __hip_bfloat16* h1s  = (__hip_bfloat16*)(smem + H1_OFF);
    __hip_bfloat16* h2s  = (__hip_bfloat16*)(smem + H2_OFF);
    float* vels  = (float*)(smem + VEL_OFF);   // [64][256]
    float* gates = (float*)(smem + GATE_OFF);  // [64][8]
    __hip_bfloat16* wst  = (__hip_bfloat16*)(smem + WST_OFF);   // 3 buffers

    const int tid   = threadIdx.x;
    const int lane  = tid & 31;
    const int wave  = tid >> 5;
    const int r0    = (wave & 3) << 4;   // 16-row block within the 64-row tile
    const int n0    = (wave >> 2) << 7;  // 128-col half for 256-wide GEMMs
    const int n0r   = (wave >> 2) << 6;  // 64-col half for 128-wide GEMMs
    const int rbase = ((lane >> 4) << 3);
    const int row0g = blockIdx.x * MTILE;

    // ---- zero velocity accumulator ----
    for (int i = tid; i < MTILE * EMBED; i += 256) vels[i] = 0.0f;

    // ---- phase 1a: z_t & cond -> x tile (bf16); sinusoidal emb (bf16) -> h1s ----
    for (int i = tid; i < MTILE * EMBED; i += 256) {
        int r = i >> 8, c = i & 255;
        xs[r * XS_LD + c]       = __float2bfloat16(z_t[(size_t)(row0g + r) * EMBED + c]);
        xs[r * XS_LD + 384 + c] = __float2bfloat16(cond[(size_t)(row0g + r) * CONDD + c]);
    }
    const float lsc = -9.2103403719761836f / 63.0f;  // -ln(10000)/(half-1)
    for (int i = tid; i < MTILE * TIMED; i += 256) {
        int r = i >> 7, c = i & 127;
        float tv  = t[row0g + r];
        float ang = tv * __expf(lsc * (float)(c & 63));
        h1s[r * H_LD + c] = __float2bfloat16((c < 64) ? __sinf(ang) : __cosf(ang));
    }
    __syncthreads();

    // ---- phase 1b: th = silu(emb @ tW1 + tb1)  (WMMA, N=128) ----
    {
        v8f acc[4];
        gemm_slice<TIMED, H_LD, 4>(h1s, r0, t1t, n0r, lane, acc);
#pragma unroll
        for (int j = 0; j < 4; ++j) {
            int   col  = n0r + 16 * j + (lane & 15);
            float bias = tb1[col];
#pragma unroll
            for (int r = 0; r < 8; ++r)
                h2s[(r0 + rbase + r) * H_LD + col] =
                    __float2bfloat16(silu_f(acc[j][r] + bias));
        }
    }
    __syncthreads();

    // ---- phase 1c: time_emb = th @ tW2 + tb2 -> x cols [256,384)  (WMMA) ----
    {
        v8f acc[4];
        gemm_slice<TIMED, H_LD, 4>(h2s, r0, t2t, n0r, lane, acc);
#pragma unroll
        for (int j = 0; j < 4; ++j) {
            int   col  = n0r + 16 * j + (lane & 15);
            float bias = tb2[col];
#pragma unroll
            for (int r = 0; r < 8; ++r)
                xs[(r0 + rbase + r) * XS_LD + 256 + col] =
                    __float2bfloat16(acc[j][r] + bias);
        }
    }
    __syncthreads();

    // ---- router hidden: rh = silu(x @ rW1 + rb1)  (WMMA, N=128) ----
    {
        v8f acc[4];
        gemm_slice<IN_DIM, XS_LD, 4>(xs, r0, r1t, n0r, lane, acc);
#pragma unroll
        for (int j = 0; j < 4; ++j) {
            int   col  = n0r + 16 * j + (lane & 15);
            float bias = rb1[col];
#pragma unroll
            for (int r = 0; r < 8; ++r)
                h1s[(r0 + rbase + r) * H_LD + col] =
                    __float2bfloat16(silu_f(acc[j][r] + bias));
        }
    }
    __syncthreads();

    // ---- router gate: logits + softmax (one thread per token) ----
    if (tid < MTILE) {
        float lg[NEXP];
#pragma unroll
        for (int e = 0; e < NEXP; ++e) lg[e] = rb2[e];
        for (int k = 0; k < RHID; ++k) {
            float rv = __bfloat162float(h1s[tid * H_LD + k]);
#pragma unroll
            for (int e = 0; e < NEXP; ++e) lg[e] += rv * rW2[k * NEXP + e];
        }
        float m = lg[0];
#pragma unroll
        for (int e = 1; e < NEXP; ++e) m = fmaxf(m, lg[e]);
        float s = 0.0f;
#pragma unroll
        for (int e = 0; e < NEXP; ++e) { lg[e] = __expf(lg[e] - m); s += lg[e]; }
        float inv = 1.0f / s;
#pragma unroll
        for (int e = 0; e < NEXP; ++e) {
            float g = lg[e] * inv;
            gates[tid * NEXP + e] = g;
            out_gate[(size_t)(row0g + tid) * NEXP + e] = g;
        }
    }
    __syncthreads();

    // ---- dense expert loop: async-staged weights, h1 -> h2 -> gated accumulate ----
    for (int e = 0; e < NEXP; ++e) {
        v8f acc[8];

        // GEMM1: h1 = silu(x @ W1_e + b1_e)
        gemm_staged<IN_DIM, XS_LD, 8>(xs, r0, w1t + (size_t)e * IN_DIM * HID,
                                      n0, tid, lane, acc, wst);
#pragma unroll
        for (int j = 0; j < 8; ++j) {
            int   col  = n0 + 16 * j + (lane & 15);
            float bias = eb1[e * HID + col];
#pragma unroll
            for (int r = 0; r < 8; ++r)
                h1s[(r0 + rbase + r) * H_LD + col] =
                    __float2bfloat16(silu_f(acc[j][r] + bias));
        }
        // gemm_staged's internal wait/barrier orders h1s writes before reads

        // GEMM2: h2 = silu(h1 @ W2_e + b2_e)
        gemm_staged<HID, H_LD, 8>(h1s, r0, w2t + (size_t)e * HID * HID,
                                  n0, tid, lane, acc, wst);
#pragma unroll
        for (int j = 0; j < 8; ++j) {
            int   col  = n0 + 16 * j + (lane & 15);
            float bias = eb2[e * HID + col];
#pragma unroll
            for (int r = 0; r < 8; ++r)
                h2s[(r0 + rbase + r) * H_LD + col] =
                    __float2bfloat16(silu_f(acc[j][r] + bias));
        }

        // GEMM3: vel += gate_e * (h2 @ W3_e + b3_e)
        gemm_staged<HID, H_LD, 8>(h2s, r0, w3t + (size_t)e * HID * EMBED,
                                  n0, tid, lane, acc, wst);
        float gg[8];
#pragma unroll
        for (int r = 0; r < 8; ++r) gg[r] = gates[(r0 + rbase + r) * NEXP + e];
#pragma unroll
        for (int j = 0; j < 8; ++j) {
            int   col  = n0 + 16 * j + (lane & 15);
            float bias = eb3[e * EMBED + col];
#pragma unroll
            for (int r = 0; r < 8; ++r)
                vels[(r0 + rbase + r) * EMBED + col] += gg[r] * (acc[j][r] + bias);
        }
        __syncthreads();
    }

    // ---- write velocity tile (vectorized) ----
    const f32x4* v4 = (const f32x4*)vels;
    f32x4* o4 = (f32x4*)out_vel + (size_t)row0g * (EMBED / 4);
    for (int i = tid; i < MTILE * EMBED / 4; i += 256) o4[i] = v4[i];
}

// ---------------- host launcher ----------------
extern "C" void kernel_launch(void* const* d_in, const int* in_sizes, int n_in,
                              void* d_out, int out_size, void* d_ws, size_t ws_size,
                              hipStream_t stream) {
    const float* z_t  = (const float*)d_in[0];
    const float* t    = (const float*)d_in[1];
    const float* cond = (const float*)d_in[2];
    const float* tW1  = (const float*)d_in[3];
    const float* tb1  = (const float*)d_in[4];
    const float* tW2  = (const float*)d_in[5];
    const float* tb2  = (const float*)d_in[6];
    const float* eW1  = (const float*)d_in[7];
    const float* eb1  = (const float*)d_in[8];
    const float* eW2  = (const float*)d_in[9];
    const float* eb2  = (const float*)d_in[10];
    const float* eW3  = (const float*)d_in[11];
    const float* eb3  = (const float*)d_in[12];
    const float* rW1  = (const float*)d_in[13];
    const float* rb1  = (const float*)d_in[14];
    const float* rW2  = (const float*)d_in[15];
    const float* rb2  = (const float*)d_in[16];

    // bf16 col-major weight workspace (~5 MB)
    __hip_bfloat16* w1t = (__hip_bfloat16*)d_ws;
    __hip_bfloat16* w2t = w1t + (size_t)NEXP * IN_DIM * HID;
    __hip_bfloat16* w3t = w2t + (size_t)NEXP * HID * HID;
    __hip_bfloat16* r1t = w3t + (size_t)NEXP * HID * EMBED;
    __hip_bfloat16* t1t = r1t + (size_t)IN_DIM * RHID;
    __hip_bfloat16* t2t = t1t + (size_t)TIMED * TIMED;

    long n1 = (long)NEXP * IN_DIM * HID;
    cvt_transpose<<<(int)((n1 + 255) / 256), 256, 0, stream>>>(eW1, w1t, IN_DIM, HID, n1);
    long n2 = (long)NEXP * HID * HID;
    cvt_transpose<<<(int)((n2 + 255) / 256), 256, 0, stream>>>(eW2, w2t, HID, HID, n2);
    cvt_transpose<<<(int)((n2 + 255) / 256), 256, 0, stream>>>(eW3, w3t, HID, EMBED, n2);
    long n4 = (long)IN_DIM * RHID;
    cvt_transpose<<<(int)((n4 + 255) / 256), 256, 0, stream>>>(rW1, r1t, IN_DIM, RHID, n4);
    long n5 = (long)TIMED * TIMED;
    cvt_transpose<<<(int)((n5 + 255) / 256), 256, 0, stream>>>(tW1, t1t, TIMED, TIMED, n5);
    cvt_transpose<<<(int)((n5 + 255) / 256), 256, 0, stream>>>(tW2, t2t, TIMED, TIMED, n5);

    float* out_vel  = (float*)d_out;
    float* out_gate = out_vel + (size_t)BTOK * EMBED;
    moe_main<<<BTOK / MTILE, 256, 0, stream>>>(z_t, t, cond, tb1, tb2,
                                               eb1, eb2, eb3, rb1, rW2, rb2,
                                               w1t, w2t, w3t, r1t, t1t, t2t,
                                               out_vel, out_gate);
}
